// DGCNN_14259291423369
// MI455X (gfx1250) — matrix operand order repaired and checked by reading, not compiled, not run on hardware
//
#include <hip/hip_runtime.h>

// ---------------- problem constants ----------------
#define B_    8
#define N_    2048
#define KNN_  21
#define P_    (B_ * N_ * KNN_)   // 344064, divisible by 256
#define P2_   (B_ * N_)          // 16384
#define EPS_  1e-5f

typedef __attribute__((ext_vector_type(16))) _Float16 v16h;
typedef __attribute__((ext_vector_type(8)))  _Float16 v8h;
typedef __attribute__((ext_vector_type(8)))  float    v8f;

union HV { v16h v16; v8h v8[2]; };

// ---------------- KNN: top-21 by neg sq dist (self included) ----------------
__global__ __launch_bounds__(256) void knn_kernel(const float* __restrict__ x,
                                                  int* __restrict__ idx) {
  __shared__ float pts[N_ * 3];
  int b = blockIdx.x >> 3;
  int nblk = blockIdx.x & 7;
  const float* xb = x + (size_t)b * N_ * 3;
  for (int i = threadIdx.x; i < N_ * 3; i += 256) pts[i] = xb[i];
  __syncthreads();
  int n = nblk * 256 + threadIdx.x;
  float px = pts[n * 3 + 0], py = pts[n * 3 + 1], pz = pts[n * 3 + 2];
  float best[KNN_]; int bid[KNN_];
#pragma unroll
  for (int j = 0; j < KNN_; ++j) { best[j] = -3.4e38f; bid[j] = 0; }
  for (int m = 0; m < N_; ++m) {
    float dx = px - pts[m * 3 + 0];
    float dy = py - pts[m * 3 + 1];
    float dz = pz - pts[m * 3 + 2];
    float d = -(dx * dx + dy * dy + dz * dz);
    if (d > best[KNN_ - 1]) {
      float cv = d; int ci = m;
#pragma unroll
      for (int j = 0; j < KNN_; ++j) {
        if (cv > best[j]) {
          float tv = best[j]; int ti = bid[j];
          best[j] = cv; bid[j] = ci; cv = tv; ci = ti;
        }
      }
    }
  }
  int* op = idx + ((size_t)b * N_ + n) * KNN_;
#pragma unroll
  for (int j = 0; j < KNN_; ++j) op[j] = bid[j];
}

// ---------------- edge features, pixel-major [p][32]: {nb-ctr(3), ctr(3), 0...} ----------------
__global__ __launch_bounds__(256) void gather_kernel(const float* __restrict__ x,
                                                     const int* __restrict__ idx,
                                                     _Float16* __restrict__ h0) {
  int p = blockIdx.x * 256 + threadIdx.x;      // < P_
  int b = p / (N_ * KNN_);
  int r = p - b * (N_ * KNN_);
  int n = r / KNN_;
  int j = idx[p];
  const float* c0 = x + ((size_t)b * N_ + n) * 3;
  const float* q0 = x + ((size_t)b * N_ + j) * 3;
  v8h o = {};
  o[0] = (_Float16)(q0[0] - c0[0]);
  o[1] = (_Float16)(q0[1] - c0[1]);
  o[2] = (_Float16)(q0[2] - c0[2]);
  o[3] = (_Float16)c0[0];
  o[4] = (_Float16)c0[1];
  o[5] = (_Float16)c0[2];
  v8h z = {};
  _Float16* row = h0 + (size_t)p * 32;
  *(v8h*)(row)      = o;
  *(v8h*)(row + 8)  = z;
  *(v8h*)(row + 16) = z;
  *(v8h*)(row + 24) = z;
}

// ---------------- weight prep: f32 [M][Kd] -> f16 [M][Kp] zero-padded ----------------
__global__ __launch_bounds__(256) void prep_weights(const float* __restrict__ W,
                                                    _Float16* __restrict__ Wh,
                                                    int Kd, int Kp) {
  int e = blockIdx.x * 256 + threadIdx.x;      // over M*Kp
  int r = e / Kp, k = e - r * Kp;
  float w = (k < Kd) ? W[(size_t)r * Kd + k] : 0.f;
  Wh[e] = (_Float16)w;
}

// ---------------- WMMA GEMM, double-buffered async-to-LDS pipeline ----------------
// Y[p][M] (f16) = A[p][Kp](f16) x Wh[M][Kp](f16)^T, f32 accumulate.
// Block tile 64(M) x 256(P), 8 waves each 32x64, K-step 32.
#define BM 64
#define BP 256
#define BK 32
__global__ __launch_bounds__(256) void gemm_f16_wmma(const _Float16* __restrict__ Wh,
                                                     const _Float16* __restrict__ A,
                                                     _Float16* __restrict__ Y,
                                                     int M, int Kp, int Pd) {
  // XOR-granule swizzle: addr(row,k) = row*32 + (((k>>3) ^ (row&3))<<3) + (k&7)
  __shared__ _Float16 Ws[2][BM * BK];   // 2 x 4 KB
  __shared__ _Float16 As[2][BP * BK];   // 2 x 16 KB
  int t = threadIdx.x;
  int lane = t & 31;
  int wid = t >> 5;
  int wm = wid & 1;                     // 32-row strip
  int wn = wid >> 1;                    // 64-col strip
  int m0 = blockIdx.x * BM;
  int p0 = blockIdx.y * BP;

  v8f acc[2][4] = {};

  // async copy of K-tile `kt` into buffer `buf` (5 x b128 per thread, no VGPR data)
  auto issue = [&](int kt, int buf) {
    int k0 = kt * BK;
    {   // weights: thread t -> row t>>2, granule t&3
      int rr = t >> 2, g = t & 3;
      const _Float16* gp = Wh + (size_t)(m0 + rr) * Kp + k0 + g * 8;
      unsigned lo = (unsigned)(uintptr_t)(&Ws[buf][rr * 32 + (((g ^ (rr & 3)) << 3))]);
      asm volatile("global_load_async_to_lds_b128 %0, %1, off"
                   :: "v"(lo), "v"(gp) : "memory");
    }
    {   // activations: thread t -> pixel-column t, granules 0..3
      const _Float16* ap = A + (size_t)(p0 + t) * Kp + k0;
#pragma unroll
      for (int g = 0; g < 4; ++g) {
        unsigned lo = (unsigned)(uintptr_t)(&As[buf][t * 32 + ((g ^ (t & 3)) << 3)]);
        asm volatile("global_load_async_to_lds_b128 %0, %1, off"
                     :: "v"(lo), "v"(ap + g * 8) : "memory");
      }
    }
  };

  int kTiles = Kp / BK;
  issue(0, 0);
  asm volatile("s_wait_asynccnt 0x0" ::: "memory");
  __syncthreads();

  for (int kt = 0; kt < kTiles; ++kt) {
    int cur = kt & 1;
    if (kt + 1 < kTiles) issue(kt + 1, cur ^ 1);   // overlap copy with compute

    // --- fragments: all b128 LDS loads ---
    int khiG = (lane >> 4);                      // granule 0/1 -> K offset 0/8
    HV af[2];
#pragma unroll
    for (int s = 0; s < 2; ++s) {
      int mrow = wm * 32 + s * 16 + (lane & 15);
      int sw = (mrow & 3);
      af[s].v8[0] = *(const v8h*)(&Ws[cur][mrow * 32 + (((khiG       ^ sw) << 3))]);
      af[s].v8[1] = *(const v8h*)(&Ws[cur][mrow * 32 + ((((khiG + 2) ^ sw) << 3))]);
    }
    int gb = (lane >> 4);                        // K base granule 0/1 -> K 0/8
    HV bf[4];
#pragma unroll
    for (int j = 0; j < 4; ++j) {
      int ncol = wn * 64 + j * 16 + (lane & 15);
      int sw = (ncol & 3);
      bf[j].v8[0] = *(const v8h*)(&As[cur][ncol * 32 + (((gb       ^ sw) << 3))]);
      bf[j].v8[1] = *(const v8h*)(&As[cur][ncol * 32 + ((((gb + 2) ^ sw) << 3))]);
    }
#pragma unroll
    for (int s = 0; s < 2; ++s)
#pragma unroll
      for (int j = 0; j < 4; ++j)
        acc[s][j] = __builtin_amdgcn_wmma_f32_16x16x32_f16(
            false, af[s].v16, false, bf[j].v16, (short)0, acc[s][j], false, false);

    if (kt + 1 < kTiles) {
      asm volatile("s_wait_asynccnt 0x0" ::: "memory");  // next buffer landed
      __syncthreads();                                   // + all waves done reading cur
    }
  }

  // --- C/D store: lane's 8 rows are M-contiguous -> one b128 per tile ---
#pragma unroll
  for (int s = 0; s < 2; ++s) {
    int mbase = m0 + wm * 32 + s * 16 + ((lane >> 4) ? 8 : 0);
#pragma unroll
    for (int j = 0; j < 4; ++j) {
      int ncol = p0 + wn * 64 + j * 16 + (lane & 15);
      v8h oh;
#pragma unroll
      for (int r = 0; r < 8; ++r) oh[r] = (_Float16)acc[s][j][r];
      *(v8h*)(Y + (size_t)ncol * M + mbase) = oh;   // global b128
    }
  }
}

// ---------------- BN statistics (pixel-major): register accum, 16 LDS atomics ----------------
__global__ void zero_stats(float* s) { s[threadIdx.x] = 0.f; }  // <<<1,1024>>>

__global__ __launch_bounds__(256) void stats_kernel(const _Float16* __restrict__ Y,
                                                    float* __restrict__ sums,
                                                    float* __restrict__ sumsq,
                                                    int M) {
  __shared__ float s1[512], s2[512];
  for (int i = threadIdx.x; i < M; i += 256) { s1[i] = 0.f; s2[i] = 0.f; }
  __syncthreads();
  // 2048 % M == 0 for all M here -> each thread's 8 channels are loop-invariant
  size_t base = (size_t)blockIdx.x * 65536 + (size_t)threadIdx.x * 8;
  float a[8] = {}, b2[8] = {};
  for (int i = 0; i < 32; ++i) {
    v8h y = *(const v8h*)(Y + base + (size_t)i * 2048);
#pragma unroll
    for (int q = 0; q < 8; ++q) {
      float v = (float)y[q];
      a[q] += v; b2[q] += v * v;
    }
  }
  int c0 = (int)((threadIdx.x * 8) & (unsigned)(M - 1));
#pragma unroll
  for (int q = 0; q < 8; ++q) {
    atomicAdd(&s1[c0 + q], a[q]);
    atomicAdd(&s2[c0 + q], b2[q]);
  }
  __syncthreads();
  for (int i = threadIdx.x; i < M; i += 256) {
    atomicAdd(&sums[i], s1[i]);
    atomicAdd(&sumsq[i], s2[i]);
  }
}

__global__ void finalize_kernel(const float* sums, const float* sumsq,
                                const float* g, const float* bt,
                                float* scale, float* shift, int M, float invCnt) {
  int c = threadIdx.x;            // <<<1,512>>>
  if (c < M) {
    float mean = sums[c] * invCnt;
    float var  = sumsq[c] * invCnt - mean * mean;   // biased, matches jnp.var
    float sc = g[c] * rsqrtf(var + EPS_);
    scale[c] = sc;
    shift[c] = bt[c] - mean * sc;
  }
}

// ---------------- BN+ReLU elementwise, pixel-major in/out, v8h vectorized ----------------
__global__ __launch_bounds__(256) void bnrelu_kernel(const _Float16* __restrict__ Y,
                                                     const float* __restrict__ scale,
                                                     const float* __restrict__ shift,
                                                     _Float16* __restrict__ Aout, int Mmask) {
  size_t e8 = ((size_t)blockIdx.x * 256 + threadIdx.x) * 8;
  v8h y = *(const v8h*)(Y + e8);
  int c0 = (int)(e8 & (size_t)Mmask);
  v8h o;
#pragma unroll
  for (int q = 0; q < 8; ++q) {
    float v = (float)y[q] * scale[c0 + q] + shift[c0 + q];
    o[q] = (_Float16)fmaxf(v, 0.f);
  }
  *(v8h*)(Aout + e8) = o;
}

// ---------------- BN+ReLU then max over K -> concat row block (cat[bn][512]) ----------------
__global__ __launch_bounds__(256) void bnrelu_max_kernel(const _Float16* __restrict__ Y,
                                                         const float* __restrict__ scale,
                                                         const float* __restrict__ shift,
                                                         _Float16* __restrict__ cat,
                                                         int cOff, int M, int log2M) {
  int lin = blockIdx.x * 256 + threadIdx.x;    // < P2_ * M
  int c = lin & (M - 1);
  int bn = lin >> log2M;
  const _Float16* yp = Y + (size_t)bn * KNN_ * M + c;
  float sc = scale[c], sh = shift[c];
  float m = 0.f;                               // relu outputs >= 0
#pragma unroll
  for (int k = 0; k < KNN_; ++k) {
    float v = (float)yp[(size_t)k * M] * sc + sh;
    m = fmaxf(m, fmaxf(v, 0.f));
  }
  cat[(size_t)bn * 512 + cOff + c] = (_Float16)m;
}

// ---------------- final BN+ReLU -> f32 out (B,512,N); coalesced writes ----------------
__global__ __launch_bounds__(256) void bnrelu_out_kernel(const _Float16* __restrict__ Y5,
                                                         const float* __restrict__ scale,
                                                         const float* __restrict__ shift,
                                                         float* __restrict__ out) {
  int c = blockIdx.y;                           // 512
  int p2 = blockIdx.x * 256 + threadIdx.x;      // 16384
  int b = p2 >> 11;
  int n = p2 & 2047;
  float v = (float)Y5[(size_t)p2 * 512 + c] * scale[c] + shift[c];
  out[((size_t)b * 512 + c) * N_ + n] = fmaxf(v, 0.f);
}

// ---------------- host orchestration ----------------
static void conv_stats(const float* Wg, _Float16* Wh, const _Float16* Ain, _Float16* Yb,
                       const float* g, const float* bt,
                       float* sums, float* sumsq, float* scale, float* shift,
                       int M, int Kd, int Kp, int Pd, hipStream_t stream) {
  prep_weights<<<(M * Kp) / 256, 256, 0, stream>>>(Wg, Wh, Kd, Kp);
  gemm_f16_wmma<<<dim3(M / BM, Pd / BP), 256, 0, stream>>>(Wh, Ain, Yb, M, Kp, Pd);
  zero_stats<<<1, 1024, 0, stream>>>(sums);
  stats_kernel<<<(int)(((size_t)Pd * M) / 65536), 256, 0, stream>>>(Yb, sums, sumsq, M);
  finalize_kernel<<<1, 512, 0, stream>>>(sums, sumsq, g, bt, scale, shift, M,
                                         1.0f / (float)Pd);
}

extern "C" void kernel_launch(void* const* d_in, const int* in_sizes, int n_in,
                              void* d_out, int out_size, void* d_ws, size_t ws_size,
                              hipStream_t stream) {
  (void)in_sizes; (void)n_in; (void)out_size; (void)ws_size;
  const float* x  = (const float*)d_in[0];
  const float* W1 = (const float*)d_in[1];
  const float* g1 = (const float*)d_in[2];
  const float* b1 = (const float*)d_in[3];
  const float* W2 = (const float*)d_in[4];
  const float* g2 = (const float*)d_in[5];
  const float* b2 = (const float*)d_in[6];
  const float* W3 = (const float*)d_in[7];
  const float* g3 = (const float*)d_in[8];
  const float* b3 = (const float*)d_in[9];
  const float* W4 = (const float*)d_in[10];
  const float* g4 = (const float*)d_in[11];
  const float* b4 = (const float*)d_in[12];
  const float* W5 = (const float*)d_in[13];
  const float* g5 = (const float*)d_in[14];
  const float* b5 = (const float*)d_in[15];
  float* out = (float*)d_out;

  char* ws = (char*)d_ws;
  size_t off = 0;
  auto take = [&](size_t bytes) -> void* {
    void* p = ws + off;
    off = (off + bytes + 255) & ~(size_t)255;
    return p;
  };
  int*       idx   = (int*)take((size_t)P_ * 4);
  _Float16*  h0    = (_Float16*)take((size_t)P_  * 32  * 2);  // [p][32], padded
  _Float16*  aA    = (_Float16*)take((size_t)P_  * 128 * 2);  // a1 [p][64], a3 [p][128]
  _Float16*  aB    = (_Float16*)take((size_t)P_  * 64  * 2);  // a2 [p][64]
  _Float16*  yb    = (_Float16*)take((size_t)P_  * 256 * 2);  // conv outputs (reused)
  _Float16*  cat   = (_Float16*)take((size_t)P2_ * 512 * 2);  // [bn][512]
  _Float16*  y5    = (_Float16*)take((size_t)P2_ * 512 * 2);
  _Float16*  Wh    = (_Float16*)take((size_t)512 * 512 * 2);  // padded f16 weights (reused)
  float*     sums  = (float*)take(1024 * 4);
  float*     sumsq = sums + 512;
  float*     scale = (float*)take(512 * 4);
  float*     shift = (float*)take(512 * 4);

  // 1) KNN + edge features
  knn_kernel<<<B_ * (N_ / 256), 256, 0, stream>>>(x, idx);
  gather_kernel<<<P_ / 256, 256, 0, stream>>>(x, idx, h0);

  // 2) stage 1: 64 <- 6 (Kp = 32)
  conv_stats(W1, Wh, h0, yb, g1, b1, sums, sumsq, scale, shift, 64, 6, 32, P_, stream);
  bnrelu_kernel<<<(int)(((size_t)P_ * 64) / 2048), 256, 0, stream>>>(yb, scale, shift, aA, 63);
  bnrelu_max_kernel<<<(P2_ * 64) / 256, 256, 0, stream>>>(yb, scale, shift, cat, 0, 64, 6);

  // 3) stage 2: 64 <- 64
  conv_stats(W2, Wh, aA, yb, g2, b2, sums, sumsq, scale, shift, 64, 64, 64, P_, stream);
  bnrelu_kernel<<<(int)(((size_t)P_ * 64) / 2048), 256, 0, stream>>>(yb, scale, shift, aB, 63);
  bnrelu_max_kernel<<<(P2_ * 64) / 256, 256, 0, stream>>>(yb, scale, shift, cat, 64, 64, 6);

  // 4) stage 3: 128 <- 64
  conv_stats(W3, Wh, aB, yb, g3, b3, sums, sumsq, scale, shift, 128, 64, 64, P_, stream);
  bnrelu_kernel<<<(int)(((size_t)P_ * 128) / 2048), 256, 0, stream>>>(yb, scale, shift, aA, 127);
  bnrelu_max_kernel<<<(P2_ * 128) / 256, 256, 0, stream>>>(yb, scale, shift, cat, 128, 128, 7);

  // 5) stage 4: 256 <- 128 (activations only feed the max)
  conv_stats(W4, Wh, aA, yb, g4, b4, sums, sumsq, scale, shift, 256, 128, 128, P_, stream);
  bnrelu_max_kernel<<<(P2_ * 256) / 256, 256, 0, stream>>>(yb, scale, shift, cat, 256, 256, 8);

  // 6) stage 5: 512 <- 512 over B*N pixels, then f32 output
  conv_stats(W5, Wh, cat, y5, g5, b5, sums, sumsq, scale, shift, 512, 512, 512, P2_, stream);
  bnrelu_out_kernel<<<dim3(P2_ / 256, 512), 256, 0, stream>>>(y5, scale, shift, out);
}
